// LTM_88965952569424
// MI455X (gfx1250) — matrix operand order repaired and checked by reading, not compiled, not run on hardware
//
#include <hip/hip_runtime.h>
#include <hip/hip_bf16.h>

// ---------------- model dims ----------------
#define DM 128
#define NH 16
#define HD 8
#define DEPTH 16
#define NV 17
#define SL 1024
#define BOS 16

#define NTH 512
#define NWAVES (NTH / 32)

// ---------------- LDS layout ----------------
#define LDS_A_BYTES 65536                 // per A buffer (double buffered)
#define LDS_B_OFF   (2 * LDS_A_BYTES)     // 131072
#define LDS_B_BYTES 131072
#define LDS_ATTN_OFF (LDS_B_OFF + LDS_B_BYTES)  // 262144; also reused as sort scratch
#define LDS_ATTN_BYTES 16384              // 16 waves x [16][32] f16
#define SMEM_BYTES (LDS_ATTN_OFF + LDS_ATTN_BYTES)  // 278528 < 320KB

// ---------------- workspace layout (bytes) ----------------
static constexpr size_t PER_LAYER_H = 384 * 128 + 128 * 128 + 512 * 128 + 128 * 512; // 196608 halves
static constexpr size_t OFF_LAT  = 0;                                   // (SL+1) ints
static constexpr size_t OFF_H    = 4352;                                // SL*DM f32
static constexpr size_t OFF_QKV  = OFF_H   + (size_t)SL * DM * 4;       // SL*384 f32
static constexpr size_t OFF_XLN  = OFF_QKV + (size_t)SL * 384 * 4;      // SL*DM f16
static constexpr size_t OFF_O16  = OFF_XLN + (size_t)SL * DM * 2;       // SL*DM f16
static constexpr size_t OFF_HID  = OFF_O16 + (size_t)SL * DM * 2;       // SL*512 f16
static constexpr size_t OFF_LOG  = OFF_HID + (size_t)SL * 512 * 2;      // SL*32 f32 (padded N)
static constexpr size_t OFF_LL   = OFF_LOG + (size_t)SL * 32 * 4;       // SL f32
static constexpr size_t OFF_MASK = OFF_LL  + (size_t)SL * 4;            // SL int
static constexpr size_t OFF_W16  = OFF_MASK + (size_t)SL * 4;           // f16 transposed weights
static constexpr size_t OFF_HEADT = OFF_W16 + (size_t)DEPTH * PER_LAYER_H * 2; // 32*128 f16
static constexpr size_t OFF_VT   = OFF_HEADT + 32 * 128 * 2;            // [128 ch][SL] f16 V transpose

typedef _Float16 v16h __attribute__((ext_vector_type(16)));
typedef _Float16 v8h  __attribute__((ext_vector_type(8)));
typedef float    v8f  __attribute__((ext_vector_type(8)));
typedef float    v2f  __attribute__((ext_vector_type(2)));

#define CAT16(lo, hi) __builtin_shufflevector(lo, hi, 0, 1, 2, 3, 4, 5, 6, 7, 8, 9, 10, 11, 12, 13, 14, 15)

struct Params {
  const int* tokens;
  const float *emb, *pos;
  const float *ln1_s, *ln1_b, *wqkv, *bqkv, *wo, *bo;
  const float *ln2_s, *ln2_b, *w1, *b1, *w2, *b2;
  const float *lnf_s, *lnf_b, *head_w, *head_b;
  int* lat;
  float* h;
  float* qkv;
  _Float16* xln;
  _Float16* o16;
  _Float16* hid16;
  _Float16* vt;
  float* logits;
  float* ll;
  int* mask;
  _Float16* w16;
  _Float16* headT;
  float* out;
};

// ---------------- feature detection ----------------
#if defined(__gfx1250__) && __has_builtin(__builtin_amdgcn_tensor_load_to_lds) && __has_builtin(__builtin_amdgcn_s_wait_tensorcnt)
#define USE_TDM 1
#endif
#if defined(__gfx1250__) && __has_builtin(__builtin_amdgcn_wmma_f32_16x16x4_f32)
#define USE_WMMA_ATTN 1
#endif

#if defined(__HIP_DEVICE_COMPILE__)
#ifdef USE_TDM
#pragma message("CDNA5-PATH: tensor_load_to_lds TDM staging ENABLED")
#else
#pragma message("CDNA5-PATH: TDM staging DISABLED (fallback block copies)")
#endif
#ifdef USE_WMMA_ATTN
#pragma message("CDNA5-PATH: wmma_f32_16x16x4_f32 attention ENABLED")
#else
#pragma message("CDNA5-PATH: WMMA attention DISABLED (VALU fallback)")
#endif
#endif

// ---------------- TDM (async tensor) staging, with fallback ----------------
#ifdef USE_TDM
typedef unsigned int u32x4 __attribute__((ext_vector_type(4)));
typedef int i32x4 __attribute__((ext_vector_type(4)));
typedef int i32x8 __attribute__((ext_vector_type(8)));

// 1-D contiguous copy, data_size = 8 bytes. Descriptor per CDNA5 ISA 08_async_tensor §8.
__device__ __forceinline__ void tdm_copy_1d(void* lds, const void* g, unsigned bytes) {
  unsigned elems = bytes >> 3; // 8-byte units
  unsigned long long ga = (unsigned long long)g;
  u32x4 g0;
  g0[0] = 1u;                                                   // count=1, user mode
  g0[1] = (unsigned)(unsigned long long)lds;                    // lds_addr
  g0[2] = (unsigned)ga;                                         // global_addr[31:0]
  g0[3] = (unsigned)((ga >> 32) & 0x1FFFFFFull) | (2u << 30);   // global_addr[56:32] + type=2
  i32x8 g1;
  g1[0] = 3 << 16;                                              // workgroup_mask=0, data_size=8B
  g1[1] = (int)((elems & 0xFFFFu) << 16);                       // tensor_dim0[15:0]
  g1[2] = (int)((elems >> 16) | (1u << 16));                    // tensor_dim0[31:16]; tensor_dim1=1
  g1[3] = (int)((elems & 0xFFFFu) << 16);                       // tile_dim0
  g1[4] = 0;                                                    // tile_dim1/2 unused
  g1[5] = (int)elems;                                           // tensor_dim0_stride
  g1[6] = 0;
  g1[7] = 0;
  i32x4 z4 = {0, 0, 0, 0};
#if __has_include(<hip/amd_detail/amd_gfx1250_TDM.h>)
  i32x8 z8 = {0, 0, 0, 0, 0, 0, 0, 0};
  __builtin_amdgcn_tensor_load_to_lds(g0, g1, z4, z4, z8, 0);   // clang-23 6-arg form
#else
  __builtin_amdgcn_tensor_load_to_lds(g0, g1, z4, z4, 0);       // ROCm 7.2 5-arg form
#endif
}
#endif

__device__ __forceinline__ void lds_copy(void* lds, const void* g, int bytes, int tid) {
#ifdef USE_TDM
  if (tid < 32) tdm_copy_1d(lds, g, (unsigned)bytes);           // wave0 issues one TDM op
#else
  int n16 = bytes >> 4;
  for (int i = tid; i < n16; i += NTH) ((int4*)lds)[i] = ((const int4*)g)[i];
#endif
}

// ---------------- WMMA GEMM: C[M,N] = A[M,K] * B^T[N,K] (+bias, gelu, resid, f16out) ----------------
// flags: 1 = gelu, 2 = residual add into Cf, 4 = f16 output to C16
__device__ void gemm_wmma(const _Float16* A, const _Float16* BT, const float* bias,
                          float* Cf, _Float16* C16, int Stiles, int N, int K,
                          int realN, int ldc, int flags, char* smem) {
  const int tid  = threadIdx.x;
  const int lane = tid & 31;
  const int wave = tid >> 5;
  const int h16  = lane >> 4;
  const int ln   = lane & 15;
  _Float16* ldsA0 = (_Float16*)smem;
  _Float16* ldsA1 = (_Float16*)(smem + LDS_A_BYTES);
  _Float16* ldsB  = (_Float16*)(smem + LDS_B_OFF);
  const int S = Stiles * 16;
  const int Ntiles = N >> 4;
  const int rowsPer = LDS_A_BYTES / (K * 2);           // 256 rows @K=128, 64 rows @K=512
  lds_copy(ldsB, BT, N * K * 2, tid);                  // whole W^T resident in LDS
  lds_copy(ldsA0, A, min(rowsPer, S) * K * 2, tid);
  int buf = 0;
  for (int m0 = 0; m0 < S; m0 += rowsPer) {
    const int rows = min(rowsPer, S - m0);
    const int nm0 = m0 + rowsPer;
    const bool pref = nm0 < S;
    _Float16* ldsA = buf ? ldsA1 : ldsA0;
    if (pref)  // prefetch next chunk into the other buffer (freed by last iteration's barrier)
      lds_copy(buf ? ldsA0 : ldsA1, A + (size_t)nm0 * K, min(rowsPer, S - nm0) * K * 2, tid);
#ifdef USE_TDM
    if (tid < 32) {  // TENSORcnt completes in order: <=1 pending leaves only the prefetch in flight
      if (pref) __builtin_amdgcn_s_wait_tensorcnt(1);
      else      __builtin_amdgcn_s_wait_tensorcnt(0);
    }
#endif
    __syncthreads();
    const int jobs = (rows >> 4) * Ntiles;
    for (int job = wave; job < jobs; job += NWAVES) {  // uniform per wave -> EXEC all ones for WMMA
      const int mt = job / Ntiles, nt = job - mt * Ntiles;
      const int col = nt * 16 + ln;
      const float bv = (col < realN) ? bias[col] : 0.f;
      v8f acc = {bv, bv, bv, bv, bv, bv, bv, bv};
      const _Float16* ap = ldsA + (size_t)(mt * 16 + ln) * K;
      const _Float16* bp = ldsB + (size_t)(nt * 16 + ln) * K;
      for (int k0 = 0; k0 < K; k0 += 32) {
        v8h alo = *(const v8h*)(ap + k0 + h16 * 8);
        v8h ahi = *(const v8h*)(ap + k0 + h16 * 8 + 16);
        v8h blo = *(const v8h*)(bp + k0 + h16 * 8);
        v8h bhi = *(const v8h*)(bp + k0 + h16 * 8 + 16);
        v16h av = CAT16(alo, ahi);
        v16h bw = CAT16(blo, bhi);
        acc = __builtin_amdgcn_wmma_f32_16x16x32_f16(false, av, false, bw, (short)0, acc, false, false);
      }
      if (col < realN) {
        const int rbase = m0 + mt * 16 + h16 * 8;
#pragma unroll
        for (int j = 0; j < 8; ++j) {
          float val = acc[j];
          if (flags & 1) {
            float x = val;
            val = 0.5f * x * (1.f + tanhf(0.7978845608f * (x + 0.044715f * x * x * x)));
          }
          size_t idx = (size_t)(rbase + j) * ldc + col;
          if (flags & 2)      Cf[idx] += val;
          else if (flags & 4) C16[idx] = (_Float16)val;
          else                Cf[idx] = val;
        }
      }
    }
    __syncthreads();
    buf ^= 1;
  }
}

// ---------------- LayerNorm (f32 in, f16 out) ----------------
__device__ void layernorm_f16(const float* src, _Float16* dst, const float* s, const float* b, int S) {
  for (int t = threadIdx.x; t < S; t += NTH) {
    const float* r = src + (size_t)t * DM;
    float m = 0.f;
    for (int d = 0; d < DM; ++d) m += r[d];
    m *= (1.f / DM);
    float v = 0.f;
    for (int d = 0; d < DM; ++d) { float x = r[d] - m; v += x * x; }
    v *= (1.f / DM);
    float inv = rsqrtf(v + 1e-5f);
    _Float16* o = dst + (size_t)t * DM;
    for (int d = 0; d < DM; ++d) o[d] = (_Float16)((r[d] - m) * inv * s[d] + b[d]);
  }
}

// ---------------- Causal attention, VALU fallback ----------------
__device__ void attention_causal(const float* qkv, _Float16* o16, int S) {
  for (int job = threadIdx.x; job < S * NH; job += NTH) {
    const int t = job >> 4, hh = job & 15;
    const float* q = qkv + (size_t)t * 384 + hh * HD;
    float m = -3.4e38f, l = 0.f, acc[HD];
#pragma unroll
    for (int d = 0; d < HD; ++d) acc[d] = 0.f;
    for (int j = 0; j <= t; ++j) {
      const float* kp = qkv + (size_t)j * 384 + 128 + hh * HD;
      const float* vp = qkv + (size_t)j * 384 + 256 + hh * HD;
      float s = 0.f;
#pragma unroll
      for (int d = 0; d < HD; ++d) s += q[d] * kp[d];
      s *= 0.35355339059327373f;
      float mn = fmaxf(m, s);
      float corr = __expf(m - mn);
      float pr = __expf(s - mn);
      l = l * corr + pr;
#pragma unroll
      for (int d = 0; d < HD; ++d) acc[d] = acc[d] * corr + pr * vp[d];
      m = mn;
    }
    float inv = 1.f / l;
    _Float16* o = o16 + (size_t)t * DM + hh * HD;
#pragma unroll
    for (int d = 0; d < HD; ++d) o[d] = (_Float16)(acc[d] * inv);
  }
}

// ---------------- Causal flash attention on WMMA ----------------
// Q.K^T via v_wmma_f32_16x16x4_f32 (2 ops cover HD=8, f32 scores), online softmax with
// 16-lane shfl_xor row reductions, P.V via v_wmma_f32_16x16x32_f16 with P restaged
// through per-wave LDS (C-layout -> A-layout) and V pre-transposed to f16 [ch][seq].
#ifdef USE_WMMA_ATTN
__device__ void attention_wmma(const float* qkv, const _Float16* Vt, _Float16* o16,
                               int Stiles, char* smem) {
  const int tid  = threadIdx.x;
  const int lane = tid & 31;
  const int wave = tid >> 5;
  const int h16  = lane >> 4;
  const int ln   = lane & 15;
  _Float16* scr = (_Float16*)(smem + LDS_ATTN_OFF) + (size_t)wave * 512; // [16][32] f16 per wave
  // zero-pad keys 16..31 of the P staging tile once (kept zero across all jobs)
#pragma unroll
  for (int j = 0; j < 8; ++j) scr[(h16 * 8 + j) * 32 + 16 + ln] = (_Float16)0.f;
  asm volatile("s_wait_dscnt 0x0" ::: "memory");
  const int njobs = Stiles * NH;
  for (int job = wave; job < njobs; job += NWAVES) {   // uniform per wave
    const int qt = job / NH, hh = job - qt * NH;
    const float* qrow = qkv + (size_t)(qt * 16 + ln) * 384 + hh * HD;
    // A-frag (16x4 f32): VGPR0 = K{0|2}, VGPR1 = K{1|3} per lane half
    v2f a0 = {qrow[2 * h16], qrow[2 * h16 + 1]};
    v2f a1 = {qrow[4 + 2 * h16], qrow[4 + 2 * h16 + 1]};
    float mrun[8], lrun[8];
    v8f oacc = {0.f, 0.f, 0.f, 0.f, 0.f, 0.f, 0.f, 0.f};
#pragma unroll
    for (int j = 0; j < 8; ++j) { mrun[j] = -3.4e38f; lrun[j] = 0.f; }
    for (int kt = 0; kt <= qt; ++kt) {
      const float* krow = qkv + (size_t)(kt * 16 + ln) * 384 + 128 + hh * HD;
      v2f b0 = {krow[2 * h16], krow[2 * h16 + 1]};
      v2f b1 = {krow[4 + 2 * h16], krow[4 + 2 * h16 + 1]};
      v8f sc = {0.f, 0.f, 0.f, 0.f, 0.f, 0.f, 0.f, 0.f};
      sc = __builtin_amdgcn_wmma_f32_16x16x4_f32(false, a0, false, b0, (short)0, sc, false, false);
      sc = __builtin_amdgcn_wmma_f32_16x16x4_f32(false, a1, false, b1, (short)0, sc, false, false);
      const int col = kt * 16 + ln;
#pragma unroll
      for (int j = 0; j < 8; ++j) {
        const int row = qt * 16 + h16 * 8 + j;
        float v = sc[j] * 0.35355339059327373f;
        if (col > row) v = -1e9f;                       // causal mask (pre-softmax, as reference)
        float tmax = v;                                 // row max across the 16 lanes of this half
        tmax = fmaxf(tmax, __shfl_xor(tmax, 8, 32));
        tmax = fmaxf(tmax, __shfl_xor(tmax, 4, 32));
        tmax = fmaxf(tmax, __shfl_xor(tmax, 2, 32));
        tmax = fmaxf(tmax, __shfl_xor(tmax, 1, 32));
        float mnew = fmaxf(mrun[j], tmax);
        float corr = __expf(mrun[j] - mnew);
        float pr = __expf(v - mnew);
        float psum = pr;                                // row sum across the half
        psum += __shfl_xor(psum, 8, 32);
        psum += __shfl_xor(psum, 4, 32);
        psum += __shfl_xor(psum, 2, 32);
        psum += __shfl_xor(psum, 1, 32);
        lrun[j] = lrun[j] * corr + psum;
        oacc[j] = oacc[j] * corr;
        mrun[j] = mnew;
        scr[(h16 * 8 + j) * 32 + ln] = (_Float16)pr;    // stage P (C-layout -> LDS row-major)
      }
      asm volatile("s_wait_dscnt 0x0" ::: "memory");    // same-wave LDS in-order: stores visible
      // reload P as a 16x32 f16 A-fragment (keys 16..31 are the zero pad)
      v8h alo = *(const v8h*)(scr + ln * 32 + h16 * 8);
      v8h ahi = *(const v8h*)(scr + ln * 32 + 16 + h16 * 8);
      v16h pa = CAT16(alo, ahi);
      // V tile as B operand: lanes (ln<8) hold keys h16*8..h16*8+7 of channel ln in elements 0..7
      v8h zro = {(_Float16)0.f, (_Float16)0.f, (_Float16)0.f, (_Float16)0.f,
                 (_Float16)0.f, (_Float16)0.f, (_Float16)0.f, (_Float16)0.f};
      v8h blo = zro, bhi = zro;
      if (ln < HD) {
        const _Float16* vp = Vt + (size_t)(hh * HD + ln) * SL + kt * 16;
        blo = *(const v8h*)(vp + h16 * 8);
      }
      v16h vb = CAT16(blo, bhi);
      oacc = __builtin_amdgcn_wmma_f32_16x16x32_f16(false, pa, false, vb, (short)0, oacc, false, false);
    }
    if (ln < HD) {
#pragma unroll
      for (int j = 0; j < 8; ++j) {
        const int row = qt * 16 + h16 * 8 + j;
        o16[(size_t)row * DM + hh * HD + ln] = (_Float16)(oacc[j] / lrun[j]);
      }
    }
  }
}
#endif

// ---------------- One transformer forward over the first Stiles*16 tokens ----------------
__device__ void forward_pass(const Params& p, int Stiles, char* smem) {
  const int tid = threadIdx.x;
  const int S = Stiles * 16;
  for (int i = tid; i < S * DM; i += NTH) {
    int t = i >> 7, d = i & 127;
    p.h[i] = p.emb[p.lat[t] * DM + d] + p.pos[i];
  }
  __syncthreads();
  for (int ly = 0; ly < DEPTH; ++ly) {
    const _Float16* wq = p.w16 + (size_t)ly * PER_LAYER_H;
    const _Float16* wo = wq + 384 * 128;
    const _Float16* w1 = wo + 128 * 128;
    const _Float16* w2 = w1 + 512 * 128;
    layernorm_f16(p.h, p.xln, p.ln1_s + ly * DM, p.ln1_b + ly * DM, S);
    __syncthreads();
    gemm_wmma(p.xln, wq, p.bqkv + ly * 384, p.qkv, nullptr, Stiles, 384, 128, 384, 384, 0, smem);
    __syncthreads();
#ifdef USE_WMMA_ATTN
    for (int i = tid; i < S * DM; i += NTH) {          // V -> f16 [channel][seq] transpose
      int t = i >> 7, c = i & 127;
      p.vt[(size_t)c * SL + t] = (_Float16)p.qkv[(size_t)t * 384 + 256 + c];
    }
    __syncthreads();
    attention_wmma(p.qkv, p.vt, p.o16, Stiles, smem);
#else
    attention_causal(p.qkv, p.o16, S);
#endif
    __syncthreads();
    gemm_wmma(p.o16, wo, p.bo + ly * DM, p.h, nullptr, Stiles, 128, 128, 128, 128, 2, smem);
    __syncthreads();
    layernorm_f16(p.h, p.xln, p.ln2_s + ly * DM, p.ln2_b + ly * DM, S);
    __syncthreads();
    gemm_wmma(p.xln, w1, p.b1 + ly * 512, nullptr, p.hid16, Stiles, 512, 128, 512, 512, 1 | 4, smem);
    __syncthreads();
    gemm_wmma(p.hid16, w2, p.b2 + ly * DM, p.h, nullptr, Stiles, 128, 512, 128, 128, 2, smem);
    __syncthreads();
  }
  layernorm_f16(p.h, p.xln, p.lnf_s, p.lnf_b, S);
  __syncthreads();
  gemm_wmma(p.xln, p.headT, p.head_b, p.logits, nullptr, Stiles, 32, 128, NV, 32, 0, smem);
  __syncthreads();
}

// ---------------- Persistent kernel: likelihood pass -> quantile mask -> heal loop -> final pass ----------------
__global__ __launch_bounds__(NTH) void ltm_tf_heal(Params p) {
  extern __shared__ char smem[];
  const int tid = threadIdx.x;

  // BOS-pad latent
  for (int i = tid; i < SL + 1; i += NTH) p.lat[i] = (i == 0) ? BOS : p.tokens[i - 1];

  // One-time f32 -> f16 W^T weight prep (weights stay hot in the 192MB L2)
  for (int ly = 0; ly < DEPTH; ++ly) {
    _Float16* wq = p.w16 + (size_t)ly * PER_LAYER_H;
    _Float16* wo = wq + 384 * 128;
    _Float16* w1 = wo + 128 * 128;
    _Float16* w2 = w1 + 512 * 128;
    const float* sq = p.wqkv + (size_t)ly * 128 * 384;
    const float* so = p.wo + (size_t)ly * 128 * 128;
    const float* s1 = p.w1 + (size_t)ly * 128 * 512;
    const float* s2 = p.w2 + (size_t)ly * 512 * 128;
    for (int i = tid; i < 384 * 128; i += NTH) { int n = i >> 7, k = i & 127; wq[i] = (_Float16)sq[k * 384 + n]; }
    for (int i = tid; i < 128 * 128; i += NTH) { int n = i >> 7, k = i & 127; wo[i] = (_Float16)so[k * 128 + n]; }
    for (int i = tid; i < 512 * 128; i += NTH) { int n = i >> 7, k = i & 127; w1[i] = (_Float16)s1[k * 512 + n]; }
    for (int i = tid; i < 128 * 512; i += NTH) { int n = i >> 9, k = i & 511; w2[i] = (_Float16)s2[k * 128 + n]; }
  }
  for (int i = tid; i < 32 * 128; i += NTH) {
    int n = i >> 7, k = i & 127;
    p.headT[i] = (n < NV) ? (_Float16)p.head_w[k * NV + n] : (_Float16)0.f;
  }
  __syncthreads();

  // Pass 0: likelihood over the original latent
  forward_pass(p, SL / 16, smem);
  for (int t = tid; t < SL; t += NTH) {
    const float* r = p.logits + (size_t)t * 32;
    float mx = r[0];
    for (int v = 1; v < NV; ++v) mx = fmaxf(mx, r[v]);
    float s = 0.f;
    for (int v = 0; v < NV; ++v) s += __expf(r[v] - mx);
    p.ll[t] = r[p.lat[t + 1]] - mx - __logf(s);
  }
  __syncthreads();

  // Bitonic sort of ll in LDS -> 3% linear-interp quantile -> heal mask
  float* sb = (float*)(smem + LDS_ATTN_OFF);
  for (int i = tid; i < SL; i += NTH) sb[i] = p.ll[i];
  __syncthreads();
  for (int k = 2; k <= SL; k <<= 1) {
    for (int j = k >> 1; j > 0; j >>= 1) {
      for (int i = tid; i < SL; i += NTH) {
        int ixj = i ^ j;
        if (ixj > i) {
          float a = sb[i], b = sb[ixj];
          bool up = ((i & k) == 0);
          if ((a > b) == up) { sb[i] = b; sb[ixj] = a; }
        }
      }
      __syncthreads();
    }
  }
  {
    float pos = 0.03f * (float)(SL - 1);
    int i0 = (int)pos;
    float fr = pos - (float)i0;
    float thr = sb[i0] + fr * (sb[i0 + 1] - sb[i0]);
    for (int i = tid; i < SL; i += NTH) p.mask[i] = (p.ll[i] < thr) ? 1 : 0;
  }
  __syncthreads();

  // Token-healing loop: causal => only the first ceil(i/16) tiles are needed per heal
  for (int i = 1; i <= SL; ++i) {
    if (p.mask[i - 1]) {
      forward_pass(p, (i + 15) >> 4, smem);
      if (tid == 0) {
        const float* r = p.logits + (size_t)(i - 1) * 32;
        int best = 0;
        float bv = r[0];
        for (int v = 1; v < BOS; ++v)  // argmax over vocab excluding BOS (=16)
          if (r[v] > bv) { bv = r[v]; best = v; }
        p.lat[i] = best;
      }
      __syncthreads();
    }
  }

  // Final pass over the healed sequence, emit outputs
  forward_pass(p, SL / 16, smem);
  for (int i = tid; i < SL * NV; i += NTH) {
    int t = i / NV, v = i - t * NV;
    p.out[i] = p.logits[(size_t)t * 32 + v];
  }
  for (int i = tid; i < SL; i += NTH) p.out[SL * NV + i] = (float)p.lat[i + 1];
}

extern "C" void kernel_launch(void* const* d_in, const int* in_sizes, int n_in,
                              void* d_out, int out_size, void* d_ws, size_t ws_size,
                              hipStream_t stream) {
  (void)in_sizes; (void)n_in; (void)out_size; (void)ws_size;
  Params p;
  p.tokens = (const int*)d_in[0];
  p.emb    = (const float*)d_in[1];
  p.pos    = (const float*)d_in[2];
  p.ln1_s  = (const float*)d_in[3];
  p.ln1_b  = (const float*)d_in[4];
  p.wqkv   = (const float*)d_in[5];
  p.bqkv   = (const float*)d_in[6];
  p.wo     = (const float*)d_in[7];
  p.bo     = (const float*)d_in[8];
  p.ln2_s  = (const float*)d_in[9];
  p.ln2_b  = (const float*)d_in[10];
  p.w1     = (const float*)d_in[11];
  p.b1     = (const float*)d_in[12];
  p.w2     = (const float*)d_in[13];
  p.b2     = (const float*)d_in[14];
  p.lnf_s  = (const float*)d_in[15];
  p.lnf_b  = (const float*)d_in[16];
  p.head_w = (const float*)d_in[17];
  p.head_b = (const float*)d_in[18];
  char* ws = (char*)d_ws;
  p.lat    = (int*)(ws + OFF_LAT);
  p.h      = (float*)(ws + OFF_H);
  p.qkv    = (float*)(ws + OFF_QKV);
  p.xln    = (_Float16*)(ws + OFF_XLN);
  p.o16    = (_Float16*)(ws + OFF_O16);
  p.hid16  = (_Float16*)(ws + OFF_HID);
  p.vt     = (_Float16*)(ws + OFF_VT);
  p.logits = (float*)(ws + OFF_LOG);
  p.ll     = (float*)(ws + OFF_LL);
  p.mask   = (int*)(ws + OFF_MASK);
  p.w16    = (_Float16*)(ws + OFF_W16);
  p.headT  = (_Float16*)(ws + OFF_HEADT);
  p.out    = (float*)d_out;
  ltm_tf_heal<<<1, NTH, SMEM_BYTES, stream>>>(p);
}
